// MHA_26216480374998
// MI455X (gfx1250) — matrix operand order repaired and checked by reading, not compiled, not run on hardware
//
#include <hip/hip_runtime.h>
#include <hip/hip_bf16.h>

typedef __attribute__((ext_vector_type(16))) _Float16 v16h;
typedef __attribute__((ext_vector_type(8)))  _Float16 v8h;
typedef __attribute__((ext_vector_type(8)))  float    v8f;
typedef __attribute__((ext_vector_type(4)))  unsigned int u32x4;
typedef __attribute__((ext_vector_type(8)))  int      i32x8;
typedef __attribute__((ext_vector_type(4)))  int      i32x4;

#define MHA_B 2
#define MHA_L 4096
#define MHA_H 8
#define MHA_D 64
#define MHA_DM 512
#define MHA_M (MHA_B * MHA_L)   // 8192 rows

#if defined(__gfx1250__) && __has_builtin(__builtin_amdgcn_tensor_load_to_lds) && \
    __has_builtin(__builtin_amdgcn_s_wait_tensorcnt)
#define USE_TDM 1
#else
#define USE_TDM 0
#endif

static __device__ __forceinline__ v16h cat16(v8h a, v8h b) {
    return __builtin_shufflevector(a, b, 0,1,2,3,4,5,6,7,8,9,10,11,12,13,14,15);
}

static __device__ __forceinline__ v8f wmma_f16(v16h a, v16h b, v8f c) {
    // D = A(16x32 f16) * B(32x16 f16) + C(16x16 f32)
    return __builtin_amdgcn_wmma_f32_16x16x32_f16(
        /*neg_a=*/false, a, /*neg_b=*/false, b,
        /*c_mod=*/(short)0, c, /*reuse_a=*/false, /*reuse_b=*/false);
}

#if USE_TDM
typedef __attribute__((address_space(3))) void lds_void_t;

static __device__ __forceinline__ unsigned lds_offset(const void* p) {
    return (unsigned)(unsigned long long)(lds_void_t*)(p);
}

// D# group 0: count=1, lds_addr, 57-bit global addr, type=2 ("image")
static __device__ __forceinline__ u32x4 tdm_g0(unsigned lds_addr,
                                               unsigned long long gaddr) {
    u32x4 g = {};
    g[0] = 1u;                                            // count=1, no gather
    g[1] = lds_addr;                                      // bits 63:32
    g[2] = (unsigned)(gaddr & 0xFFFFFFFFu);               // bits 95:64
    g[3] = (unsigned)((gaddr >> 32) & 0x01FFFFFFu) | (2u << 30); // addr hi | type=2
    return g;
}

// D# group 1: data_size=2B, tensor/tile dims + dim0/dim1 strides (2D tile)
static __device__ __forceinline__ i32x8 tdm_g1(unsigned td0, unsigned td1,
                                               unsigned tile0, unsigned tile1,
                                               unsigned long long stride0,
                                               unsigned long long stride1) {
    i32x8 g = {};
    g[0] = (int)(1u << 16);                               // data_size=1 -> 2 bytes
    g[1] = (int)((td0 & 0xFFFFu) << 16);                  // tensor_dim0[15:0] @ 63:48
    g[2] = (int)(((td0 >> 16) & 0xFFFFu) | ((td1 & 0xFFFFu) << 16));
    g[3] = (int)(((td1 >> 16) & 0xFFFFu) | ((tile0 & 0xFFFFu) << 16));
    g[4] = (int)(tile1 & 0xFFFFu);                        // tile_dim1; tile_dim2=0
    g[5] = (int)(unsigned)(stride0 & 0xFFFFFFFFu);        // stride0[31:0]
    g[6] = (int)(((stride0 >> 32) & 0xFFFFu) | ((stride1 & 0xFFFFu) << 16));
    g[7] = (int)((unsigned)((stride1 >> 16) & 0xFFFFFFFFu));
    return g;
}
#endif

// ---------------------------------------------------------------------------
// fp32 -> f16 elementwise convert (grid-stride)
// ---------------------------------------------------------------------------
__global__ void cvt_f16_kernel(const float* __restrict__ in,
                               _Float16* __restrict__ out, int n) {
    int i = blockIdx.x * blockDim.x + threadIdx.x;
    int stride = gridDim.x * blockDim.x;
    for (; i < n; i += stride) out[i] = (_Float16)in[i];
}

// ---------------------------------------------------------------------------
// fp32 [R][C] -> f16 transposed [C][R]   (weights: row-major KxN -> NxK)
// ---------------------------------------------------------------------------
__global__ void cvt_t_f16_kernel(const float* __restrict__ in,
                                 _Float16* __restrict__ out, int R, int C) {
    int i = blockIdx.x * blockDim.x + threadIdx.x;
    int stride = gridDim.x * blockDim.x;
    int n = R * C;
    for (; i < n; i += stride) {
        int r = i / C, c = i - r * C;
        out[(size_t)c * R + r] = (_Float16)in[i];
    }
}

// ---------------------------------------------------------------------------
// WMMA GEMM:  C[m][n] = scale * sum_k A[m][k] * Bt[n][k]
//   A  : f16 row-major [M][K]
//   Bt : f16 [N][K] (B transposed -> B-fragments are contiguous 32B loads)
// Output modes:
//   0 : f16 head-major [B][H][L][64]      (Q, K projections)
//   1 : f16 d-major    [B][H][64][L]      (V projection -> V^T, feeds TDM tiles)
//   2 : f32 row-major  [M][N]             (final output projection)
// Block = 256 threads (8 waves); each wave computes a 16(M) x 64(N) strip.
// ---------------------------------------------------------------------------
__global__ __launch_bounds__(256)
void wmma_gemm_kernel(const _Float16* __restrict__ A,
                      const _Float16* __restrict__ Bt,
                      _Float16* __restrict__ outH,
                      float* __restrict__ outF32,
                      int M, int N, int K, int Lseq, float scale, int mode) {
    const int lane = threadIdx.x & 31;
    const int wave = threadIdx.x >> 5;
    const int l16  = lane & 15;
    const int half = lane >> 4;

    const int m0 = blockIdx.x * 128 + wave * 16;   // 8 waves * 16 rows
    const int n0 = blockIdx.y * 64;

    v8f acc[4] = {};

    const _Float16* arow = A + (size_t)(m0 + l16) * K;

#pragma unroll 4
    for (int kc = 0; kc < K; kc += 32) {
        __builtin_prefetch(arow + kc + 256, 0, 3);   // near-scope prefetch ahead
        // A fragment: row = l16; elems e -> k = kc + 16*half + e
        v8h a0 = *(const v8h*)(arow + kc + 16 * half);
        v8h a1 = *(const v8h*)(arow + kc + 16 * half + 8);
        v16h af = cat16(a0, a1);
#pragma unroll
        for (int j = 0; j < 4; ++j) {
            // B fragment: col = n0+16j+l16; elems e -> k = kc + 16*half + e
            const v16h bf =
                *(const v16h*)(Bt + (size_t)(n0 + 16 * j + l16) * K + kc + half * 16);
            acc[j] = wmma_f16(af, bf, acc[j]);
        }
    }

#pragma unroll
    for (int j = 0; j < 4; ++j) {
#pragma unroll
        for (int r = 0; r < 8; ++r) {
            int m = m0 + r + 8 * half;       // C layout: M = r + 8*(lane>=16)
            int n = n0 + 16 * j + l16;       // N = lane & 15 within tile
            float v = acc[j][r] * scale;
            if (mode == 2) {
                outF32[(size_t)m * N + n] = v;
            } else {
                int b = m / Lseq, l = m - b * Lseq;
                int h = n >> 6, d = n & 63;
                size_t idx = (mode == 0)
                    ? (((size_t)b * MHA_H + h) * Lseq + l) * MHA_D + d
                    : (((size_t)b * MHA_H + h) * MHA_D + d) * Lseq + l;
                outH[idx] = (_Float16)v;
            }
        }
    }
}

// ---------------------------------------------------------------------------
// Flash attention over one (b,h): online-softmax. K and V^T tiles staged into
// double-buffered LDS by the Tensor Data Mover, one tile ahead of compute
// (s_wait_tensorcnt 2 drains only the older pair). Fallback: vector copies.
// Block = 128 threads (4 waves), each wave owns 16 query rows.
// gridDim = (L/64, B*H).
// ---------------------------------------------------------------------------
__global__ __launch_bounds__(128)
void flash_attn_kernel(const _Float16* __restrict__ Qh,   // [B*H][L][64], pre-scaled
                       const _Float16* __restrict__ Kh,   // [B*H][L][64]
                       const _Float16* __restrict__ Vt,   // [B*H][64][L]  (V^T)
                       const int* __restrict__ amask,     // [B][L]
                       _Float16* __restrict__ AO) {       // [B*L][512]
    __shared__ __align__(32) _Float16 ldsK [2][64 * 64];  // K tiles  [key][d]
    __shared__ __align__(32) _Float16 ldsVt[2][64 * 64];  // V^T tiles [d][key]
    __shared__ __align__(32) _Float16 ldsP [4 * 16 * 64]; // per-wave P tile [m][key]
    __shared__ float ldsBias[64];

    const int tid  = threadIdx.x;
    const int lane = tid & 31;
    const int wave = tid >> 5;
    const int l16  = lane & 15;
    const int half = lane >> 4;

    const int bh = blockIdx.y;
    const int b  = bh >> 3;          // H == 8
    const int h  = bh & 7;
    const int q0 = blockIdx.x * 64 + wave * 16;

    const _Float16* Q   = Qh + (size_t)bh * MHA_L * MHA_D;
    const _Float16* Kp  = Kh + (size_t)bh * MHA_L * MHA_D;
    const _Float16* Vtg = Vt + (size_t)bh * MHA_D * MHA_L;

    // Q A-fragments for d-chunks {0..31},{32..63}; reused over all KV tiles.
    v16h qfrag[2];
    {
        const _Float16* qrow = Q + (size_t)(q0 + l16) * MHA_D;
#pragma unroll
        for (int c = 0; c < 2; ++c) {
            v8h a0 = *(const v8h*)(qrow + 32 * c + 16 * half);
            v8h a1 = *(const v8h*)(qrow + 32 * c + 16 * half + 8);
            qfrag[c] = cat16(a0, a1);
        }
    }

    v8f o[4] = {};
    float mstat[8], lstat[8];
#pragma unroll
    for (int r = 0; r < 8; ++r) { mstat[r] = -1e30f; lstat[r] = 0.0f; }

    const int NT = MHA_L / 64;

#if USE_TDM
    // TDM issue for one 64-key tile into LDS buffer `buf` (2 DMA descriptors).
    auto tdm_issue = [&](int kt, int buf) {
        // K tile: 64 consecutive rows of 64 halves = one contiguous 1-D run.
        const unsigned long long kg =
            (unsigned long long)(const void*)(Kp + (size_t)kt * 64 * MHA_D);
        u32x4 g0k = tdm_g0(lds_offset(&ldsK[buf][0]), kg);
        i32x8 g1k = tdm_g1(/*td0=*/4096, /*td1=*/1, /*tile0=*/4096, /*tile1=*/0,
                           /*stride0=*/4096ull, /*stride1=*/0ull);
        // V^T tile: 64 rows x 64 halves, row stride = L (d-major layout).
        const unsigned long long vg =
            (unsigned long long)(const void*)(Vtg + kt * 64);
        u32x4 g0v = tdm_g0(lds_offset(&ldsVt[buf][0]), vg);
        i32x8 g1v = tdm_g1(/*td0=*/64, /*td1=*/64, /*tile0=*/64, /*tile1=*/64,
                           /*stride0=*/(unsigned long long)MHA_L, /*stride1=*/0ull);
        i32x4 z4 = {};
        i32x8 z8 = {};
        __builtin_amdgcn_tensor_load_to_lds(g0k, g1k, z4, z4, z8, 0);
        __builtin_amdgcn_tensor_load_to_lds(g0v, g1v, z4, z4, z8, 0);
    };
    if (wave == 0) tdm_issue(0, 0);      // prologue: tile 0 in flight
#endif

    for (int kt = 0; kt < NT; ++kt) {
        const int kbase = kt * 64;
#if USE_TDM
        const int cur = kt & 1;
        // All waves done reading buf[cur^1] (iteration kt-1) before the DMA
        // below overwrites it with tile kt+1.
        __syncthreads();
        if (wave == 0) {
            if (kt + 1 < NT) {
                tdm_issue(kt + 1, cur ^ 1);
                __builtin_amdgcn_s_wait_tensorcnt(2);  // drain tile kt's pair only
            } else {
                __builtin_amdgcn_s_wait_tensorcnt(0);
            }
        }
#else
        const int cur = 0;
        __syncthreads();
        for (int i = tid; i < 64 * 8; i += 128) {      // K tile, 16B chunks
            *(v8h*)(&ldsK[0][0] + i * 8) =
                *(const v8h*)(Kp + (size_t)kbase * MHA_D + i * 8);
        }
        for (int i = tid; i < 64 * 8; i += 128) {      // V^T tile, 16B chunks
            int d = i >> 3, seg = i & 7;
            *(v8h*)(&ldsVt[0][0] + d * 64 + seg * 8) =
                *(const v8h*)(Vtg + (size_t)d * MHA_L + kbase + seg * 8);
        }
#endif
        if (tid < 64)
            ldsBias[tid] = (amask[(size_t)b * MHA_L + kbase + tid] == 0) ? -1e30f : 0.0f;
        __syncthreads();

        const _Float16* kbuf = &ldsK[cur][0];
        const _Float16* vbuf = &ldsVt[cur][0];

        // S = Q * K^T  (4 column tiles of 16 keys, d-dim = 64 in 2 chunks)
        v8f s[4];
#pragma unroll
        for (int j = 0; j < 4; ++j) {
            v8f acc = {};
#pragma unroll
            for (int c = 0; c < 2; ++c) {
                // B-frag: col = key 16j+l16; elems e -> d = 32c + 16*half + e
                const v16h bf =
                    *(const v16h*)(kbuf + (16 * j + l16) * 64 + 32 * c + half * 16);
                acc = wmma_f16(qfrag[c], bf, acc);
            }
            acc += ldsBias[16 * j + l16];   // per-key-column mask bias
            s[j] = acc;
        }

        // Online softmax update (per row; 16-lane reductions within wave half)
#pragma unroll
        for (int r = 0; r < 8; ++r) {
            float tm = fmaxf(fmaxf(s[0][r], s[1][r]), fmaxf(s[2][r], s[3][r]));
#pragma unroll
            for (int ofs = 1; ofs < 16; ofs <<= 1)
                tm = fmaxf(tm, __shfl_xor(tm, ofs, 32));
            float mn = fmaxf(mstat[r], tm);
            float alpha = __expf(mstat[r] - mn);
            mstat[r] = mn;

            float rs = 0.0f;
#pragma unroll
            for (int j = 0; j < 4; ++j) {
                float p = __expf(s[j][r] - mn);
                s[j][r] = p;
                rs += p;
            }
#pragma unroll
            for (int ofs = 1; ofs < 16; ofs <<= 1)
                rs += __shfl_xor(rs, ofs, 32);
            lstat[r] = lstat[r] * alpha + rs;
#pragma unroll
            for (int j = 0; j < 4; ++j) o[j][r] *= alpha;
        }

        // Write P (C-layout) into per-wave LDS tile as f16, row-major [m][key].
        _Float16* pw = ldsP + wave * 16 * 64;
#pragma unroll
        for (int j = 0; j < 4; ++j)
#pragma unroll
            for (int r = 0; r < 8; ++r)
                pw[(r + 8 * half) * 64 + 16 * j + l16] = (_Float16)s[j][r];
        // LDS ops are in-order within a wave: loads below see the stores above.

        // O += P * V   (A-frags of P from LDS, B-frags from V^T tile)
#pragma unroll
        for (int c = 0; c < 2; ++c) {
            const _Float16* prow = pw + l16 * 64 + 32 * c;
            v8h p0 = *(const v8h*)(prow + 16 * half);
            v8h p1 = *(const v8h*)(prow + 16 * half + 8);
            v16h pa = cat16(p0, p1);
#pragma unroll
            for (int j = 0; j < 4; ++j) {
                const v16h bf =
                    *(const v16h*)(vbuf + (16 * j + l16) * 64 + 32 * c + half * 16);
                o[j] = wmma_f16(pa, bf, o[j]);
            }
        }
    }

    // Normalize and write attention output row-major [B*L][H*64] as f16.
#pragma unroll
    for (int r = 0; r < 8; ++r) {
        float inv = 1.0f / lstat[r];
        int m = q0 + r + 8 * half;
#pragma unroll
        for (int j = 0; j < 4; ++j) {
            AO[(size_t)(b * MHA_L + m) * MHA_DM + h * MHA_D + 16 * j + l16] =
                (_Float16)(o[j][r] * inv);
        }
    }
}

// ---------------------------------------------------------------------------
// Host launcher
// ---------------------------------------------------------------------------
extern "C" void kernel_launch(void* const* d_in, const int* in_sizes, int n_in,
                              void* d_out, int out_size, void* d_ws, size_t ws_size,
                              hipStream_t stream) {
    (void)in_sizes; (void)n_in; (void)out_size; (void)ws_size;

    const float* q  = (const float*)d_in[0];
    const float* k  = (const float*)d_in[1];
    const float* v  = (const float*)d_in[2];
    const int*   am = (const int*)d_in[3];
    const float* Wq = (const float*)d_in[4];
    const float* Wk = (const float*)d_in[5];
    const float* Wv = (const float*)d_in[6];
    const float* Wo = (const float*)d_in[7];

    constexpr size_t SZ_X16 = (size_t)MHA_M * MHA_DM * sizeof(_Float16);  // 8 MB
    constexpr size_t SZ_W16 = (size_t)MHA_DM * MHA_DM * sizeof(_Float16); // 512 KB

    char* ws = (char*)d_ws;
    _Float16* xq16 = (_Float16*)(ws);
    _Float16* xk16 = (_Float16*)(ws + SZ_X16);
    _Float16* xv16 = (_Float16*)(ws + 2 * SZ_X16);
    _Float16* wtq  = (_Float16*)(ws + 3 * SZ_X16);
    _Float16* wtk  = (_Float16*)(ws + 3 * SZ_X16 + SZ_W16);
    _Float16* wtv  = (_Float16*)(ws + 3 * SZ_X16 + 2 * SZ_W16);
    _Float16* wto  = (_Float16*)(ws + 3 * SZ_X16 + 3 * SZ_W16);
    _Float16* qh   = (_Float16*)(ws + 3 * SZ_X16 + 4 * SZ_W16);
    _Float16* kh   = (_Float16*)(ws + 4 * SZ_X16 + 4 * SZ_W16);
    _Float16* vht  = (_Float16*)(ws + 5 * SZ_X16 + 4 * SZ_W16);
    _Float16* ao   = (_Float16*)(ws + 6 * SZ_X16 + 4 * SZ_W16);

    const int nX = MHA_M * MHA_DM;          // 4,194,304
    cvt_f16_kernel<<<2048, 256, 0, stream>>>(q, xq16, nX);
    cvt_f16_kernel<<<2048, 256, 0, stream>>>(k, xk16, nX);
    cvt_f16_kernel<<<2048, 256, 0, stream>>>(v, xv16, nX);

    cvt_t_f16_kernel<<<1024, 256, 0, stream>>>(Wq, wtq, MHA_DM, MHA_DM);
    cvt_t_f16_kernel<<<1024, 256, 0, stream>>>(Wk, wtk, MHA_DM, MHA_DM);
    cvt_t_f16_kernel<<<1024, 256, 0, stream>>>(Wv, wtv, MHA_DM, MHA_DM);
    cvt_t_f16_kernel<<<1024, 256, 0, stream>>>(Wo, wto, MHA_DM, MHA_DM);

    dim3 gg(MHA_M / 128, MHA_DM / 64);      // (64, 8)
    const float qscale = 0.125f;            // 1/sqrt(64) folded into Q projection
    wmma_gemm_kernel<<<gg, 256, 0, stream>>>(xq16, wtq, qh, nullptr,
                                             MHA_M, MHA_DM, MHA_DM, MHA_L, qscale, 0);
    wmma_gemm_kernel<<<gg, 256, 0, stream>>>(xk16, wtk, kh, nullptr,
                                             MHA_M, MHA_DM, MHA_DM, MHA_L, 1.0f, 0);
    wmma_gemm_kernel<<<gg, 256, 0, stream>>>(xv16, wtv, vht, nullptr,
                                             MHA_M, MHA_DM, MHA_DM, MHA_L, 1.0f, 1);

    flash_attn_kernel<<<dim3(MHA_L / 64, MHA_B * MHA_H), 128, 0, stream>>>(
        qh, kh, vht, am, ao);

    wmma_gemm_kernel<<<gg, 256, 0, stream>>>(ao, wto, nullptr, (float*)d_out,
                                             MHA_M, MHA_DM, MHA_DM, MHA_L, 1.0f, 2);
}
